// GNN_1778116460569
// MI455X (gfx1250) — matrix operand order repaired and checked by reading, not compiled, not run on hardware
//
#include <hip/hip_runtime.h>

#define NN 100000
#define NE 800000
#define EMB 256
#define HID 512
#define NLAYER 5
#define BN_EPS 1e-5f

typedef __attribute__((ext_vector_type(16))) __bf16 v16bf;
typedef __attribute__((ext_vector_type(8)))  float  v8f;

struct U4x2 { uint4 lo, hi; };
struct US4  { unsigned short a, b, c, d; };

__device__ __forceinline__ unsigned short f2bf(float f) {
  unsigned int u = __builtin_bit_cast(unsigned int, f);
  u += 0x7FFFu + ((u >> 16) & 1u);   // round-to-nearest-even
  return (unsigned short)(u >> 16);
}

__device__ __forceinline__ v16bf frag_cast(uint4 lo, uint4 hi) {
  U4x2 t{lo, hi};
  return __builtin_bit_cast(v16bf, t);
}

// ---------------- node embedding: h = x_emb1[atom] + x_emb2[chir] ----------
__global__ void embed_kernel(const int* __restrict__ xa, const int* __restrict__ xc,
                             const float* __restrict__ e1, const float* __restrict__ e2,
                             float* __restrict__ h, int n) {
  int idx = blockIdx.x * blockDim.x + threadIdx.x;
  if (idx >= n * 64) return;
  int i = idx >> 6, cc = (idx & 63) << 2;
  float4 a = *reinterpret_cast<const float4*>(e1 + (size_t)xa[i] * EMB + cc);
  float4 b = *reinterpret_cast<const float4*>(e2 + (size_t)xc[i] * EMB + cc);
  float4 o{a.x + b.x, a.y + b.y, a.z + b.z, a.w + b.w};
  *reinterpret_cast<float4*>(h + (size_t)i * EMB + cc) = o;
}

// ------------- agg = h + ee(self loop: type 4, dir 0)  (folds self loops) --
__global__ void agg_init(const float* __restrict__ h, const float* __restrict__ e1s,
                         const float* __restrict__ e2s, float* __restrict__ agg, int n) {
  int idx = blockIdx.x * blockDim.x + threadIdx.x;
  if (idx >= n * 64) return;
  int i = idx >> 6, cc = (idx & 63) << 2;
  float4 a = *reinterpret_cast<const float4*>(h + (size_t)i * EMB + cc);
  float4 b = *reinterpret_cast<const float4*>(e1s + cc);
  float4 c = *reinterpret_cast<const float4*>(e2s + cc);
  float4 o{a.x + b.x + c.x, a.y + b.y + c.y, a.z + b.z + c.z, a.w + b.w + c.w};
  *reinterpret_cast<float4*>(agg + (size_t)i * EMB + cc) = o;
}

// ------------- edge scatter: agg[dst] += h[src] + ee   (one wave / edge) ---
__global__ void scatter_edges(const float* __restrict__ h, const int* __restrict__ src,
                              const int* __restrict__ dst, const int* __restrict__ et,
                              const int* __restrict__ ed, const float* __restrict__ e1,
                              const float* __restrict__ e2, float* __restrict__ agg, int E) {
  int lane = threadIdx.x & 31;
  int e = blockIdx.x * (blockDim.x >> 5) + (threadIdx.x >> 5);
  if (e >= E) return;
  const float* hs = h + (size_t)src[e] * EMB;
  const float* t1 = e1 + (size_t)et[e] * EMB;
  const float* t2 = e2 + (size_t)ed[e] * EMB;
  float* ad = agg + (size_t)dst[e] * EMB;
#pragma unroll
  for (int k = 0; k < 8; k++) {
    int c = lane + k * 32;
    atomicAdd(&ad[c], hs[c] + t1[c] + t2[c]);
  }
}

// ------------- pack weights fp32 -> bf16 in WMMA B-fragment order ----------
// B 32x16 bf16 packing used here (contiguous k, same as A frag): lane L -> n = L%16, half = L/16; element j -> K = 16*half + j
__global__ void pack_w(const float* __restrict__ W, unsigned short* __restrict__ P,
                       int cols, int total) {
  int idx = blockIdx.x * blockDim.x + threadIdx.x;
  if (idx >= total) return;
  int j = idx & 15;
  int lane = (idx >> 4) & 31;
  int tile = idx >> 9;
  int ctn = cols >> 4;
  int ct = tile % ctn;
  int kt = tile / ctn;
  int k = kt * 32 + (lane >> 4) * 16 + j;
  int c = ct * 16 + (lane & 15);
  P[idx] = f2bf(W[(size_t)k * cols + c]);
}

// ------------------------- tiled WMMA GEMM ---------------------------------
// MODE 0: out = relu(A@B + bias) stored bf16.  MODE 1: out fp32 + BN stat atomics.
template <typename AT, int K, int COLS, int WAVES_M, int WAVES_N, int MODE>
__global__ __launch_bounds__(WAVES_M* WAVES_N * 32) void gemm_wmma(
    const AT* __restrict__ A, const unsigned short* __restrict__ Bp,
    const float* __restrict__ bias, void* __restrict__ outv,
    float* __restrict__ bn_sum, float* __restrict__ bn_sumsq, int nrows) {
  constexpr int M_BLK = WAVES_M * 16;
  constexpr int LDA = K + 8;                       // pad -> conflict-free b128 reads
  constexpr int THREADS = WAVES_M * WAVES_N * 32;
  constexpr int CPW = COLS / WAVES_N;
  constexpr int CHUNKS = CPW / 64;
  constexpr int CT = COLS / 16;

  __shared__ unsigned short ldsA[M_BLK * LDA];

  const int tid = threadIdx.x;
  const int lane = tid & 31;
  const int wave = tid >> 5;
  const int wm = wave % WAVES_M;
  const int wn = wave / WAVES_M;
  const int row0 = blockIdx.x * M_BLK;

  if constexpr (sizeof(AT) == 4) {                 // fp32 source -> convert
    constexpr int CPR = K / 4;
    for (int c = tid; c < M_BLK * CPR; c += THREADS) {
      int r = c / CPR, cc = (c % CPR) * 4;
      int gr = row0 + r;
      float4 v = make_float4(0.f, 0.f, 0.f, 0.f);
      if (gr < nrows)
        v = *reinterpret_cast<const float4*>((const float*)A + (size_t)gr * K + cc);
      US4 o{f2bf(v.x), f2bf(v.y), f2bf(v.z), f2bf(v.w)};
      *reinterpret_cast<US4*>(&ldsA[r * LDA + cc]) = o;
    }
  } else {                                         // already bf16
    constexpr int CPR = K / 8;
    for (int c = tid; c < M_BLK * CPR; c += THREADS) {
      int r = c / CPR, cc = (c % CPR) * 8;
      int gr = row0 + r;
      uint4 v = make_uint4(0u, 0u, 0u, 0u);
      if (gr < nrows)
        v = *reinterpret_cast<const uint4*>((const unsigned short*)A + (size_t)gr * K + cc);
      *reinterpret_cast<uint4*>(&ldsA[r * LDA + cc]) = v;
    }
  }
  __syncthreads();

  const int half = lane >> 4;
  const unsigned short* aRow = &ldsA[(wm * 16 + (lane & 15)) * LDA + half * 16];

  for (int ch = 0; ch < CHUNKS; ch++) {
    const int ct0 = wn * (CPW / 16) + ch * 4;
    v8f acc[4] = {};
    for (int kk = 0; kk < K; kk += 32) {
      // A frag: element j at K = kk+16*half+j
      uint4 alo = *reinterpret_cast<const uint4*>(aRow + kk);
      uint4 ahi = *reinterpret_cast<const uint4*>(aRow + kk + 8);
      v16bf af = frag_cast(alo, ahi);
      const unsigned short* bb =
          Bp + (((size_t)(kk >> 5) * CT + ct0) * 32 + lane) * 16;
#pragma unroll
      for (int t = 0; t < 4; t++) {
        uint4 blo = *reinterpret_cast<const uint4*>(bb + t * 512);
        uint4 bhi = *reinterpret_cast<const uint4*>(bb + t * 512 + 8);
        v16bf bf = frag_cast(blo, bhi);
        acc[t] = __builtin_amdgcn_wmma_f32_16x16x32_bf16(
            false, af, false, bf, (short)0, acc[t], false, false);
      }
    }
    const int n = lane & 15;
#pragma unroll
    for (int t = 0; t < 4; t++) {
      const int col = (ct0 + t) * 16 + n;
      const float bv = bias[col];
      if constexpr (MODE == 0) {
        unsigned short* ob = (unsigned short*)outv;
#pragma unroll
        for (int r = 0; r < 8; r++) {
          int gr = row0 + wm * 16 + half * 8 + r;
          if (gr < nrows) {
            float v = acc[t][r] + bv;
            v = fmaxf(v, 0.f);
            ob[(size_t)gr * COLS + col] = f2bf(v);
          }
        }
      } else {
        float* of = (float*)outv;
        float s = 0.f, sq = 0.f;
#pragma unroll
        for (int r = 0; r < 8; r++) {
          int gr = row0 + wm * 16 + half * 8 + r;
          if (gr < nrows) {
            float v = acc[t][r] + bv;
            of[(size_t)gr * COLS + col] = v;
            s += v;
            sq += v * v;
          }
        }
        atomicAdd(&bn_sum[col], s);
        atomicAdd(&bn_sumsq[col], sq);
      }
    }
  }
}

// ------------------------- BatchNorm helpers -------------------------------
__global__ void bn_zero(float* __restrict__ bn) {
  bn[threadIdx.x] = 0.f;
  bn[threadIdx.x + 256] = 0.f;
}

__global__ void bn_finalize(const float* __restrict__ bn, const float* __restrict__ gamma,
                            const float* __restrict__ beta, float* __restrict__ scale,
                            float* __restrict__ shift, int n) {
  int c = threadIdx.x;
  float mean = bn[c] / (float)n;
  float var = bn[256 + c] / (float)n - mean * mean;
  float sc = gamma[c] * rsqrtf(var + BN_EPS);
  scale[c] = sc;
  shift[c] = beta[c] - mean * sc;
}

__global__ void bn_apply(const float* __restrict__ hl, const float* __restrict__ scale,
                         const float* __restrict__ shift, float* __restrict__ outp,
                         int relu, int n) {
  int idx = blockIdx.x * blockDim.x + threadIdx.x;
  if (idx >= n * 64) return;
  int i = idx >> 6, cc = (idx & 63) << 2;
  float4 v = *reinterpret_cast<const float4*>(hl + (size_t)i * EMB + cc);
  float4 sc = *reinterpret_cast<const float4*>(scale + cc);
  float4 sh = *reinterpret_cast<const float4*>(shift + cc);
  float4 o{v.x * sc.x + sh.x, v.y * sc.y + sh.y, v.z * sc.z + sh.z, v.w * sc.w + sh.w};
  if (relu) {
    o.x = fmaxf(o.x, 0.f); o.y = fmaxf(o.y, 0.f);
    o.z = fmaxf(o.z, 0.f); o.w = fmaxf(o.w, 0.f);
  }
  *reinterpret_cast<float4*>(outp + (size_t)i * EMB + cc) = o;
}

// ---------------------------------------------------------------------------
extern "C" void kernel_launch(void* const* d_in, const int* in_sizes, int n_in,
                              void* d_out, int out_size, void* d_ws, size_t ws_size,
                              hipStream_t stream) {
  const int* x_atom = (const int*)d_in[0];
  const int* x_chir = (const int*)d_in[1];
  const int* esrc = (const int*)d_in[2];
  const int* edst = (const int*)d_in[3];
  const int* etyp = (const int*)d_in[4];
  const int* edir = (const int*)d_in[5];
  const float* x_emb1 = (const float*)d_in[6];
  const float* x_emb2 = (const float*)d_in[7];
  const float* eemb1 = (const float*)d_in[8];
  const float* eemb2 = (const float*)d_in[9];
  const float* W1 = (const float*)d_in[10];
  const float* b1 = (const float*)d_in[11];
  const float* W2 = (const float*)d_in[12];
  const float* b2 = (const float*)d_in[13];
  const float* gamma = (const float*)d_in[14];
  const float* beta = (const float*)d_in[15];

  const int N = NN, E = NE;
  float* h = (float*)d_ws;                                   // N*256 f32
  float* agg = h + (size_t)N * EMB;                          // N*256 f32 (also hl)
  unsigned short* hidden = (unsigned short*)(agg + (size_t)N * EMB);  // N*512 bf16
  unsigned short* w1p = hidden + (size_t)N * HID;            // 256*512 bf16 packed
  unsigned short* w2p = w1p + EMB * HID;                     // 512*256 bf16 packed
  float* bn = (float*)(w2p + HID * EMB);                     // sum|sumsq|scale|shift

  const int gElem = (N * 64 + 255) / 256;

  embed_kernel<<<gElem, 256, 0, stream>>>(x_atom, x_chir, x_emb1, x_emb2, h, N);

  for (int l = 0; l < NLAYER; l++) {
    pack_w<<<(EMB * HID + 255) / 256, 256, 0, stream>>>(
        W1 + (size_t)l * EMB * HID, w1p, HID, EMB * HID);
    pack_w<<<(HID * EMB + 255) / 256, 256, 0, stream>>>(
        W2 + (size_t)l * HID * EMB, w2p, EMB, HID * EMB);
    bn_zero<<<1, 256, 0, stream>>>(bn);
    agg_init<<<gElem, 256, 0, stream>>>(
        h, eemb1 + ((size_t)l * 6 + 4) * EMB, eemb2 + ((size_t)l * 3 + 0) * EMB, agg, N);
    scatter_edges<<<(E + 7) / 8, 256, 0, stream>>>(
        h, esrc, edst, etyp, edir,
        eemb1 + (size_t)l * 6 * EMB, eemb2 + (size_t)l * 3 * EMB, agg, E);
    gemm_wmma<float, EMB, HID, 6, 1, 0><<<(N + 95) / 96, 192, 0, stream>>>(
        agg, w1p, b1 + (size_t)l * HID, hidden, nullptr, nullptr, N);
    gemm_wmma<unsigned short, HID, EMB, 3, 2, 1><<<(N + 47) / 48, 192, 0, stream>>>(
        hidden, w2p, b2 + (size_t)l * EMB, agg, bn, bn + 256, N);
    bn_finalize<<<1, 256, 0, stream>>>(
        bn, gamma + (size_t)l * EMB, beta + (size_t)l * EMB, bn + 512, bn + 768, N);
    float* outp = (l == NLAYER - 1) ? (float*)d_out : h;
    bn_apply<<<gElem, 256, 0, stream>>>(agg, bn + 512, bn + 768, outp,
                                        (l < NLAYER - 1) ? 1 : 0, N);
  }
}
